// BasicBlock_50689204027962
// MI455X (gfx1250) — compile-verified, matrix operand order checked by reading, not run on hardware
//
#include <hip/hip_runtime.h>

// ---------------------------------------------------------------------------
// BasicBlock (conv3x3 -> SE gate w/ global quantile prune -> BN -> relu ->
// conv3x3 -> BN -> +residual -> relu) for gfx1250 (MI455X), wave32 WMMA.
//
// Shapes: B=8192, C=64, H=W=8 (IMG = 4096 elems/image).
// Convs done as implicit GEMM per image: M=64 spatial, N=64 c_out, K=576
// (tap-major: K = tap*64 + ci), on v_wmma_f32_16x16x32_bf16, with the input
// image tile staged into LDS by the Tensor Data Mover (TENSOR_LOAD_TO_LDS).
// ---------------------------------------------------------------------------

typedef __bf16 bf16_t;
typedef bf16_t v16bf __attribute__((ext_vector_type(16)));
typedef float  v8f   __attribute__((ext_vector_type(8)));
typedef unsigned int u32x4 __attribute__((ext_vector_type(4)));
typedef int          i32x4 __attribute__((ext_vector_type(4)));
typedef int          i32x8 __attribute__((ext_vector_type(8)));

union Frag16 { v16bf v; uint4 q[2]; };

#define NB      8192
#define NC      64
#define IMG     4096          // 64 spatial * 64 ch
#define NGATE   (NB * NC)     // 524288
#define K_RANK  104857u       // int(0.2 * 524288)
#define BN_N    524288.0f     // B*H*W

__device__ __forceinline__ unsigned short f2bf(float f) {
    unsigned int u = __float_as_uint(f);
    unsigned int r = (u + 0x7FFFu + ((u >> 16) & 1u)) >> 16;
    return (unsigned short)r;
}

// ---------------------------------------------------------------------------
// TDM: DMA one bf16 [64][64] image tile (8KB) from global to LDS.
// D# group0/group1 per CDNA5 ISA 8.3/8.4:
//   g0: count=1 | lds_addr | global_addr[31:0] | addr[56:32] + type=2
//   g1: data_size=2B; tensor_dim0/1=64; tile_dim0/1=64; dim0_stride=64
// 2-D tensor -> groups 2/3 are NULL/zero.
// ---------------------------------------------------------------------------
#if defined(__has_builtin)
#if __has_builtin(__builtin_amdgcn_tensor_load_to_lds)
#define HAVE_TDM 1
#endif
#endif

__device__ __forceinline__ void tdm_load_image(const unsigned short* gptr, unsigned lds_off) {
#ifdef HAVE_TDM
    unsigned long long ga = (unsigned long long)(size_t)gptr;
    u32x4 g0;
    g0[0] = 1u;                                        // count=1 (valid descriptor)
    g0[1] = lds_off;                                   // lds_addr (bytes)
    g0[2] = (unsigned)(ga & 0xFFFFFFFFu);              // global_addr[31:0]
    g0[3] = (unsigned)((ga >> 32) & 0x01FFFFFFu)       // global_addr[56:32]
          | 0x80000000u;                               // type=2 ("image")
    i32x8 g1;
    g1[0] = 0x00010000;       // wg_mask=0, data_size=1 (2 bytes)
    g1[1] = 64 << 16;         // tensor_dim0[15:0] = 64   (bits 63:48)
    g1[2] = 64 << 16;         // tensor_dim1[15:0] = 64   (bits 111:80 -> low half)
    g1[3] = 64 << 16;         // tile_dim0 = 64           (bits 127:112)
    g1[4] = 64;               // tile_dim1 = 64           (bits 143:128)
    g1[5] = 64;               // tensor_dim0_stride = 64  (bits 207:160)
    g1[6] = 0;
    g1[7] = 0;
    i32x4 z4 = {0, 0, 0, 0};
#if __clang_major__ >= 23
    i32x8 z8 = {0, 0, 0, 0, 0, 0, 0, 0};
    __builtin_amdgcn_tensor_load_to_lds(g0, g1, z4, z4, z8, 0);
#else
    __builtin_amdgcn_tensor_load_to_lds(g0, g1, z4, z4, 0);
#endif
#else
    (void)gptr; (void)lds_off;
#endif
}

// ---------------------------------------------------------------------------
// 1) Weight prep: wT[o][tap*64+ci] = bf16(w[o][ci][tap])   (both convs)
// ---------------------------------------------------------------------------
__global__ void prep_weights(const float* __restrict__ w1, const float* __restrict__ w2,
                             unsigned short* __restrict__ wT1, unsigned short* __restrict__ wT2) {
    int idx = blockIdx.x * 256 + threadIdx.x;
    if (idx >= 2 * 36864) return;
    const float* w = (idx < 36864) ? w1 : w2;
    unsigned short* o = (idx < 36864) ? wT1 : wT2;
    int r = idx % 36864;
    int oc = r / 576, K = r % 576;
    int tap = K >> 6, ci = K & 63;
    o[r] = f2bf(w[(oc * 64 + ci) * 9 + tap]);
}

// ---------------------------------------------------------------------------
// 2) Per image: xT(bf16, [s][c]) + channel avg-pool (SE input)
// ---------------------------------------------------------------------------
__global__ void pool_transpose(const float* __restrict__ x, unsigned short* __restrict__ xT,
                               float* __restrict__ pooled) {
    __shared__ unsigned short tr[IMG];
    __shared__ float pr[NC];
    int b = blockIdx.x, t = threadIdx.x;
    if (t < NC) pr[t] = 0.f;
    __syncthreads();
    const float* xi = x + (size_t)b * IMG;
    for (int rep = 0; rep < 4; ++rep) {
        int off = rep * 1024 + t * 4;          // off = c*64 + s
        float4 v = *(const float4*)(xi + off);
        int c = off >> 6, s = off & 63;
        tr[(s + 0) * 64 + c] = f2bf(v.x);
        tr[(s + 1) * 64 + c] = f2bf(v.y);
        tr[(s + 2) * 64 + c] = f2bf(v.z);
        tr[(s + 3) * 64 + c] = f2bf(v.w);
        atomicAdd(&pr[c], v.x + v.y + v.z + v.w);
    }
    __syncthreads();
    unsigned short* xo = xT + (size_t)b * IMG;
    for (int rep = 0; rep < 4; ++rep) {
        int off = rep * 1024 + t * 4;
        uint2 d;
        d.x = ((unsigned)tr[off + 1] << 16) | tr[off + 0];
        d.y = ((unsigned)tr[off + 3] << 16) | tr[off + 2];
        *(uint2*)(xo + off) = d;
    }
    if (t < NC) pooled[(size_t)b * NC + t] = pr[t] * (1.0f / 64.0f);
}

// ---------------------------------------------------------------------------
// 3) SE MLP: fc1(64->16) relu, fc2(16->64) sigmoid. One wave per image.
// ---------------------------------------------------------------------------
__global__ void gate_kernel(const float* __restrict__ pooled,
                            const float* __restrict__ fc1w, const float* __restrict__ fc1b,
                            const float* __restrict__ fc2w, const float* __restrict__ fc2b,
                            float* __restrict__ gates) {
    __shared__ float sp[8][NC];
    __shared__ float sh[8][16];
    int t = threadIdx.x, wave = t >> 5, lane = t & 31;
    int b = blockIdx.x * 8 + wave;
    const float* pb = pooled + (size_t)b * NC;
    sp[wave][lane]      = pb[lane];
    sp[wave][lane + 32] = pb[lane + 32];
    __syncthreads();
    if (lane < 16) {
        float h = fc1b[lane];
        const float* wr = fc1w + lane * 64;
        for (int i = 0; i < 64; ++i) h += sp[wave][i] * wr[i];
        sh[wave][lane] = h > 0.f ? h : 0.f;
    }
    __syncthreads();
    for (int half = 0; half < 2; ++half) {
        int c = lane + half * 32;
        float g = fc2b[c];
        const float* wr = fc2w + c * 16;
        for (int j = 0; j < 16; ++j) g += sh[wave][j] * wr[j];
        g = 1.0f / (1.0f + __expf(-g));
        gates[(size_t)b * NC + c] = g;
    }
}

// ---------------------------------------------------------------------------
// 4) Exact k-th smallest via 2-level radix select on float bits
// ---------------------------------------------------------------------------
__global__ void zero_u32(unsigned int* p, int n) {
    int i = blockIdx.x * 256 + threadIdx.x;
    if (i < n) p[i] = 0u;
}

__global__ void hist_hi(const float* __restrict__ g, unsigned int* __restrict__ hist) {
    int i = blockIdx.x * 256 + threadIdx.x;
    unsigned u = __float_as_uint(g[i]);
    atomicAdd(&hist[u >> 16], 1u);
}

__global__ void hist_lo(const float* __restrict__ g, const unsigned int* __restrict__ selBin,
                        unsigned int* __restrict__ hist) {
    int i = blockIdx.x * 256 + threadIdx.x;
    unsigned u = __float_as_uint(g[i]);
    if ((u >> 16) == selBin[0]) atomicAdd(&hist[u & 0xFFFFu], 1u);
}

__global__ void select_rank(const unsigned int* hist, unsigned int* selBin,
                            unsigned int* rankRem, float* thrOut, int phase) {
    __shared__ unsigned int csum[256];
    int t = threadIdx.x;
    unsigned s = 0;
    const unsigned int* hp = hist + t * 256;
    for (int i = 0; i < 256; ++i) s += hp[i];
    csum[t] = s;
    __syncthreads();
    if (t == 0) {
        unsigned k = (phase == 0) ? K_RANK : rankRem[0];
        unsigned cum = 0; int chunk = 0;
        while (chunk < 255 && cum + csum[chunk] <= k) { cum += csum[chunk]; ++chunk; }
        int bin = chunk * 256;
        while ((bin & 255) != 255 && cum + hist[bin] <= k) { cum += hist[bin]; ++bin; }
        if (phase == 0) { selBin[0] = (unsigned)bin; rankRem[0] = k - cum; }
        else            { thrOut[0] = __uint_as_float((selBin[0] << 16) | (unsigned)bin); }
    }
}

// ---------------------------------------------------------------------------
// 5) WMMA implicit-GEMM conv3x3 (shared by conv1/conv2).
//    Block = 256 thr = 8 waves = 2 images x 4 ntiles. Wave owns 4 M-tiles.
//    Images staged to LDS by TDM; A-fragments read from LDS (ds_load_b128),
//    B (weight) fragments from global (L2-broadcast).
// ---------------------------------------------------------------------------
__global__ void __launch_bounds__(256)
conv_wmma(const unsigned short* __restrict__ inT,
          const unsigned short* __restrict__ wT,
          float* __restrict__ out,
          const float* __restrict__ gates, const float* __restrict__ thrp,
          float* __restrict__ ssum, float* __restrict__ ssq) {
    __shared__ unsigned short simg[2 * IMG];   // 2 images x 8KB bf16 [s][c]
    __shared__ float redS[NC];
    __shared__ float redQ[NC];
    int tid = threadIdx.x, wave = tid >> 5, lane = tid & 31;
    if (tid < NC) { redS[tid] = 0.f; redQ[tid] = 0.f; }

    int slot = wave >> 2;                      // 0 or 1: image within block
    int img  = blockIdx.x * 2 + slot;
    const unsigned short* xim = inT + (size_t)img * IMG;

#ifdef HAVE_TDM
    // One TDM issue per image (waves 0 and 4); EXEC is ignored by tensor ops,
    // the branch is wave-uniform so only those waves issue.
    if ((wave & 3) == 0) {
        unsigned lds_off = (unsigned)(size_t)(&simg[slot * IMG]);  // low 32 = LDS byte offset
        tdm_load_image(xim, lds_off);
    }
    __builtin_amdgcn_s_wait_tensorcnt(0);
#else
    // Fallback: cooperative copy, 128 threads per image, 64B each.
    {
        int ct = tid & 127;
        const uint4* src = (const uint4*)(inT + (size_t)(blockIdx.x * 2 + (tid >> 7)) * IMG);
        uint4* dst = (uint4*)(&simg[(tid >> 7) * IMG]);
        dst[ct]       = src[ct];
        dst[ct + 128] = src[ct + 128];
        dst[ct + 256] = src[ct + 256];
        dst[ct + 384] = src[ct + 384];
    }
#endif
    __syncthreads();

    int nt  = wave & 3;
    int n0  = lane & 15;            // N col for B/D, M row for A
    int kb  = (lane >> 4) << 3;     // K sub-base per half-wave (0 or 8)
    int cOut = nt * 16 + n0;
    const unsigned short* wrow = wT + cOut * 576;
    const unsigned short* lim  = &simg[slot * IMG];

    v8f acc[4];
    #pragma unroll
    for (int i = 0; i < 4; ++i)
        #pragma unroll
        for (int j = 0; j < 8; ++j) acc[i][j] = 0.f;

    for (int kk = 0; kk < 18; ++kk) {           // K = 18 * 32, one tap per 2 steps
        int tap = kk >> 1;
        int cb  = ((kk & 1) << 5) + kb;         // input-channel base of this slice
        int dh  = tap / 3 - 1, dw = tap % 3 - 1;

        Frag16 bfr;                              // B: weights [K][N], N = cOut
        bfr.q[0] = *(const uint4*)(wrow + (kk << 5) + kb);
        bfr.q[1] = *(const uint4*)(wrow + (kk << 5) + kb + 16);

        #pragma unroll
        for (int mt = 0; mt < 4; ++mt) {
            int s  = (mt << 4) + n0;
            int hh = (s >> 3) + dh, ww = (s & 7) + dw;
            Frag16 afr;                          // A: im2col row from LDS
            if (((unsigned)hh < 8u) && ((unsigned)ww < 8u)) {
                const unsigned short* p = lim + ((((hh << 3) + ww)) << 6) + cb;
                afr.q[0] = *(const uint4*)(p);
                afr.q[1] = *(const uint4*)(p + 16);
            } else {
                afr.q[0] = make_uint4(0, 0, 0, 0);
                afr.q[1] = make_uint4(0, 0, 0, 0);
            }
            acc[mt] = __builtin_amdgcn_wmma_f32_16x16x32_bf16(
                false, afr.v, false, bfr.v, (short)0, acc[mt], false, false);
        }
    }

    float g = 1.0f;
    if (gates) {                                  // SE gate w/ global prune threshold
        g = gates[(size_t)img * NC + cOut] - thrp[0];
        g = g > 0.f ? g : 0.f;
    }

    float lsum = 0.f, lsq = 0.f;
    int sOff = (lane < 16) ? 0 : 8;               // D layout: VGPR r -> M row
    #pragma unroll
    for (int mt = 0; mt < 4; ++mt) {
        #pragma unroll
        for (int r = 0; r < 8; ++r) {
            float v = acc[mt][r] * g;
            out[(size_t)img * IMG + ((mt << 4) + sOff + r) * 64 + cOut] = v;
            lsum += v; lsq += v * v;
        }
    }
    // lanes L and L^16 hold the same channel -> fold, then LDS, then global
    lsum += __shfl_xor(lsum, 16, 32);
    lsq  += __shfl_xor(lsq, 16, 32);
    if (lane < 16) {
        atomicAdd(&redS[cOut], lsum);
        atomicAdd(&redQ[cOut], lsq);
    }
    __syncthreads();
    if (tid < NC) {
        atomicAdd(&ssum[tid], redS[tid]);
        atomicAdd(&ssq[tid], redQ[tid]);
    }
}

// ---------------------------------------------------------------------------
// 6) BN finalize: scale/bias per channel from batch sums
// ---------------------------------------------------------------------------
__global__ void finalize_stats(const float* __restrict__ sum, const float* __restrict__ sq,
                               const float* __restrict__ gamma, const float* __restrict__ beta,
                               float* __restrict__ scale, float* __restrict__ bias) {
    int c = threadIdx.x;
    const float inv_n = 1.0f / BN_N;
    float m = sum[c] * inv_n;
    float var = sq[c] * inv_n - m * m;
    float inv = rsqrtf(var + 1e-5f);
    float sc = gamma[c] * inv;
    scale[c] = sc;
    bias[c] = beta[c] - m * sc;
}

// ---------------------------------------------------------------------------
// 7) z = bf16(relu(bn1(y)))  (elementwise, [b][s][c] layout, 4 elems/thread)
// ---------------------------------------------------------------------------
__global__ void bn_relu_bf16(const float* __restrict__ y, const float* __restrict__ scale,
                             const float* __restrict__ bias, unsigned short* __restrict__ z) {
    size_t i = ((size_t)blockIdx.x * 256 + threadIdx.x) * 4;
    float4 v = *(const float4*)(y + i);
    int c0 = (int)(i & 63);
    float r0 = v.x * scale[c0 + 0] + bias[c0 + 0]; r0 = r0 > 0.f ? r0 : 0.f;
    float r1 = v.y * scale[c0 + 1] + bias[c0 + 1]; r1 = r1 > 0.f ? r1 : 0.f;
    float r2 = v.z * scale[c0 + 2] + bias[c0 + 2]; r2 = r2 > 0.f ? r2 : 0.f;
    float r3 = v.w * scale[c0 + 3] + bias[c0 + 3]; r3 = r3 > 0.f ? r3 : 0.f;
    uint2 d;
    d.x = ((unsigned)f2bf(r1) << 16) | f2bf(r0);
    d.y = ((unsigned)f2bf(r3) << 16) | f2bf(r2);
    *(uint2*)(z + i) = d;
}

// ---------------------------------------------------------------------------
// 8) Epilogue: out[c][s] = relu(bn2(u[s][c]) + x[c][s]); in-place on d_out
// ---------------------------------------------------------------------------
__global__ void epilogue_kernel(float* __restrict__ out, const float* __restrict__ x,
                                const float* __restrict__ scale, const float* __restrict__ bias) {
    __shared__ float sh[IMG];
    int b = blockIdx.x, t = threadIdx.x;
    float* ub = out + (size_t)b * IMG;
    const float* xb = x + (size_t)b * IMG;
    for (int rep = 0; rep < 4; ++rep) {
        int off = rep * 1024 + t * 4;           // [s][c] linear
        *(float4*)(sh + off) = *(const float4*)(ub + off);
    }
    __syncthreads();
    for (int rep = 0; rep < 4; ++rep) {
        int j = rep * 1024 + t * 4;             // j = c*64 + s
        int c = j >> 6, s0 = j & 63;
        float sc = scale[c], bi = bias[c];
        float4 xr = *(const float4*)(xb + j);
        float4 o;
        o.x = sh[(s0 + 0) * 64 + c] * sc + bi + xr.x; o.x = o.x > 0.f ? o.x : 0.f;
        o.y = sh[(s0 + 1) * 64 + c] * sc + bi + xr.y; o.y = o.y > 0.f ? o.y : 0.f;
        o.z = sh[(s0 + 2) * 64 + c] * sc + bi + xr.z; o.z = o.z > 0.f ? o.z : 0.f;
        o.w = sh[(s0 + 3) * 64 + c] * sc + bi + xr.w; o.w = o.w > 0.f ? o.w : 0.f;
        *(float4*)(ub + j) = o;
    }
}

// ---------------------------------------------------------------------------
extern "C" void kernel_launch(void* const* d_in, const int* in_sizes, int n_in,
                              void* d_out, int out_size, void* d_ws, size_t ws_size,
                              hipStream_t stream) {
    (void)in_sizes; (void)n_in; (void)out_size; (void)ws_size;
    const float* x    = (const float*)d_in[0];
    const float* w1   = (const float*)d_in[1];
    const float* w2   = (const float*)d_in[2];
    const float* fc1w = (const float*)d_in[3];
    const float* fc1b = (const float*)d_in[4];
    const float* fc2w = (const float*)d_in[5];
    const float* fc2b = (const float*)d_in[6];
    const float* bn1g = (const float*)d_in[7];
    const float* bn1b = (const float*)d_in[8];
    const float* bn2g = (const float*)d_in[9];
    const float* bn2b = (const float*)d_in[10];
    float* out = (float*)d_out;

    char* ws = (char*)d_ws;
    size_t off = 0;
    unsigned short* xT  = (unsigned short*)(ws + off); off += (size_t)NB * IMG * 2;  // bf16 [b][s][c]; reused as z
    float*          y   = (float*)(ws + off);          off += (size_t)NB * IMG * 4;  // conv1 gated out
    unsigned short* wT1 = (unsigned short*)(ws + off); off += 36864 * 2;
    unsigned short* wT2 = (unsigned short*)(ws + off); off += 36864 * 2;
    float*        pooled = (float*)(ws + off);         off += (size_t)NGATE * 4;
    float*        gates  = (float*)(ws + off);         off += (size_t)NGATE * 4;
    unsigned int* hist1  = (unsigned int*)(ws + off);  off += 65536 * 4;
    unsigned int* hist2  = (unsigned int*)(ws + off);  off += 65536 * 4;
    float* sum1   = (float*)(ws + off); off += NC * 4;
    float* sq1    = (float*)(ws + off); off += NC * 4;
    float* sum2   = (float*)(ws + off); off += NC * 4;
    float* sq2    = (float*)(ws + off); off += NC * 4;
    float* scale1 = (float*)(ws + off); off += NC * 4;
    float* bias1  = (float*)(ws + off); off += NC * 4;
    float* scale2 = (float*)(ws + off); off += NC * 4;
    float* bias2  = (float*)(ws + off); off += NC * 4;
    unsigned int* selBin  = (unsigned int*)(ws + off); off += 4;
    unsigned int* rankRem = (unsigned int*)(ws + off); off += 4;
    float*        thr     = (float*)(ws + off);        off += 4;

    int nz = 2 * 65536 + 8 * NC + 3;
    zero_u32<<<(nz + 255) / 256, 256, 0, stream>>>(hist1, nz);

    prep_weights<<<288, 256, 0, stream>>>(w1, w2, wT1, wT2);
    pool_transpose<<<NB, 256, 0, stream>>>(x, xT, pooled);
    gate_kernel<<<NB / 8, 256, 0, stream>>>(pooled, fc1w, fc1b, fc2w, fc2b, gates);

    hist_hi<<<NGATE / 256, 256, 0, stream>>>(gates, hist1);
    select_rank<<<1, 256, 0, stream>>>(hist1, selBin, rankRem, thr, 0);
    hist_lo<<<NGATE / 256, 256, 0, stream>>>(gates, selBin, hist2);
    select_rank<<<1, 256, 0, stream>>>(hist2, selBin, rankRem, thr, 1);

    // conv1 (gated) -> y, BN1 stats
    conv_wmma<<<NB / 2, 256, 0, stream>>>(xT, wT1, y, gates, thr, sum1, sq1);
    finalize_stats<<<1, NC, 0, stream>>>(sum1, sq1, bn1g, bn1b, scale1, bias1);

    // z = bf16(relu(bn1(y))), reusing xT storage
    bn_relu_bf16<<<(NB * IMG) / 1024, 256, 0, stream>>>(y, scale1, bias1, xT);

    // conv2 -> d_out (as [b][s][c] scratch), BN2 stats
    conv_wmma<<<NB / 2, 256, 0, stream>>>(xT, wT2, out, nullptr, thr, sum2, sq2);
    finalize_stats<<<1, NC, 0, stream>>>(sum2, sq2, bn2g, bn2b, scale2, bias2);

    // out = relu(bn2(u) + x), transposed back to NCHW in place
    epilogue_kernel<<<NB, 256, 0, stream>>>(out, x, scale2, bias2);
}
